// QuantumProjection_31765578121535
// MI455X (gfx1250) — matrix operand order repaired and checked by reading, not compiled, not run on hardware
//
#include <hip/hip_runtime.h>

typedef __attribute__((ext_vector_type(16))) _Float16 v16h;
typedef __attribute__((ext_vector_type(8)))  float    v8f;
typedef __attribute__((ext_vector_type(2)))  float    v2f;

#define INV_SQRT2 0.70710678118654752440f
#define PI_F      3.14159265358979323846f

// Generic complex 2x2 gate applied to qubit with bit-mask m (compile-time after unroll).
__device__ __forceinline__ void apply2x2(float* sr, float* si, const int m,
                                         const float u00r, const float u00i,
                                         const float u01r, const float u01i,
                                         const float u10r, const float u10i,
                                         const float u11r, const float u11i)
{
    #pragma unroll
    for (int b = 0; b < 16; ++b) {
        if (b & m) continue;
        const int bb = b | m;
        const float r0 = sr[b], i0 = si[b], r1 = sr[bb], i1 = si[bb];
        sr[b]  = u00r * r0 - u00i * i0 + u01r * r1 - u01i * i1;
        si[b]  = u00r * i0 + u00i * r0 + u01r * i1 + u01i * r1;
        sr[bb] = u10r * r0 - u10i * i0 + u11r * r1 - u11i * i1;
        si[bb] = u10r * i0 + u10i * r0 + u11r * i1 + u11i * r1;
    }
}

// Fused: f = tanh(x@W1^T + b1)*pi  ->  4-qubit circuit  ->  out = q@W2^T + b2
// Block = 256 threads (8 wave32). Each wave handles 32 rows (2 WMMA tiles).
__global__ __launch_bounds__(256) void qp_fused_kernel(
    const float* __restrict__ x,    // (B,256)
    const float* __restrict__ W1,   // (4,256)
    const float* __restrict__ b1,   // (4,)
    const float* __restrict__ Wq,   // (2,4,3)
    const float* __restrict__ W2,   // (64,4)
    const float* __restrict__ b2,   // (64,)
    float* __restrict__ out)        // (B,64)
{
    __shared__ float lds_f[256 * 4];
    __shared__ float lds_q[256 * 4];
    __shared__ float lds_rot[8 * 4];   // per Rot gate: {ct*cA, ct*sA, st*cB, st*sB}

    const int tid  = threadIdx.x;
    const int lane = tid & 31;
    const int wv   = tid >> 5;          // wave id 0..7
    const int half = (lane >> 4) & 1;   // lane half
    const int l16  = lane & 15;
    const long rowBlock = (long)blockIdx.x * 256;

    // ---- Precompute batch-constant Rot(phi,theta,omega) factors (8 gates, 8 threads) ----
    if (tid < 8) {
        const float phi = Wq[tid * 3 + 0];
        const float th  = Wq[tid * 3 + 1];
        const float om  = Wq[tid * 3 + 2];
        const float ct = __cosf(th * 0.5f), st = __sinf(th * 0.5f);
        const float A  = (phi + om) * 0.5f, Bt = (phi - om) * 0.5f;
        lds_rot[tid * 4 + 0] = ct * __cosf(A);
        lds_rot[tid * 4 + 1] = ct * __sinf(A);
        lds_rot[tid * 4 + 2] = st * __cosf(Bt);
        lds_rot[tid * 4 + 3] = st * __sinf(Bt);
    }

    // =============== Phase 1: GEMM1 via v_wmma_f32_16x16x32_f16 ===============
    v8f acc0 = {};
    v8f acc1 = {};
    const long rowT0 = rowBlock + (long)wv * 32 + l16;
    const long rowT1 = rowT0 + 16;

    #pragma unroll
    for (int kb = 0; kb < 8; ++kb) {
        const int kbase = kb * 32;

        // B operand (32x16 f16): lane -> N=l16; halves 0..7 -> K = kbase + half*8 + h, 8..15 -> K = kbase + half*8 + 16 + (h-8)
        v16h bmat;
        if (l16 < 4) {
            const float* wrow = W1 + l16 * 256 + kbase + half * 8;
            #pragma unroll
            for (int h = 0; h < 16; ++h) bmat[h] = (_Float16)wrow[h < 8 ? h : h + 8];
        } else {
            #pragma unroll
            for (int h = 0; h < 16; ++h) bmat[h] = (_Float16)0.0f;
        }

        // A operand (16x32 f16): lane -> M=l16; halves 0..7 -> K=klo+h, 8..15 -> K=klo+16+h
        const int klo = kbase + half * 8;
        v16h a0, a1;
        {
            const float* p0 = x + rowT0 * 256 + klo;
            const float* p1 = x + rowT1 * 256 + klo;
            #pragma unroll
            for (int h = 0; h < 8; ++h) {
                a0[h]     = (_Float16)p0[h];
                a0[h + 8] = (_Float16)p0[h + 16];
                a1[h]     = (_Float16)p1[h];
                a1[h + 8] = (_Float16)p1[h + 16];
            }
        }

        acc0 = __builtin_amdgcn_wmma_f32_16x16x32_f16(false, a0, false, bmat, (short)0, acc0, false, false);
        acc1 = __builtin_amdgcn_wmma_f32_16x16x32_f16(false, a1, false, bmat, (short)0, acc1, false, false);
    }

    // C layout: VGPR r, lane -> N = l16, M = r + 8*half. Keep only N=0..3.
    if (l16 < 4) {
        #pragma unroll
        for (int r = 0; r < 8; ++r) {
            const int m = r + 8 * half;
            lds_f[(wv * 32 + m) * 4 + l16]      = acc0[r];
            lds_f[(wv * 32 + 16 + m) * 4 + l16] = acc1[r];
        }
    }
    __syncthreads();

    // =============== Phase 2: 4-qubit circuit, one row per thread ===============
    {
        const int row = tid;
        float sr[16], si[16];
        #pragma unroll
        for (int b = 0; b < 16; ++b) { sr[b] = 0.0f; si[b] = 0.0f; }
        sr[0] = 1.0f;

        // Per wire: fused U = RZ(rz) * RY(ry) * H, with
        //   ry = arctan(f), rz = arctan(f^2), f = pi*tanh(acc + b1)
        // Half-angle trig of arctan(u) computed algebraically (no atan/sin/cos):
        //   cos(arctan u) = rsqrt(1+u^2); cos(t/2)=sqrt((1+c)/2); sin(t/2)=u*c/(2*cos(t/2))
        #pragma unroll
        for (int w = 0; w < 4; ++w) {
            const float v   = lds_f[row * 4 + w] + b1[w];
            const float e   = __expf(2.0f * v);
            const float tnh = 1.0f - 2.0f * __builtin_amdgcn_rcpf(e + 1.0f);
            const float f   = PI_F * tnh;

            const float c1 = __builtin_amdgcn_rsqf(1.0f + f * f);    // cos(arctan f)
            const float h1 = 0.5f * (1.0f + c1);
            const float r1 = __builtin_amdgcn_rsqf(h1);
            const float cy = h1 * r1;                                // cos(ry/2)
            const float sy = 0.5f * f * c1 * r1;                     // sin(ry/2)

            const float g  = f * f;
            const float c2 = __builtin_amdgcn_rsqf(1.0f + g * g);    // cos(arctan f^2)
            const float h2 = 0.5f * (1.0f + c2);
            const float r2 = __builtin_amdgcn_rsqf(h2);
            const float cz = h2 * r2;                                // cos(rz/2)
            const float sz = 0.5f * g * c2 * r2;                     // sin(rz/2)

            // RY*H = [[cy-sy, cy+sy],[cy+sy, sy-cy]]/sqrt2 ; then RZ phases rows.
            const float p = (cy - sy) * INV_SQRT2;
            const float q = (cy + sy) * INV_SQRT2;
            apply2x2(sr, si, 8 >> w,
                     p * cz, -p * sz,    // u00
                     q * cz, -q * sz,    // u01
                     q * cz,  q * sz,    // u10
                    -p * cz, -p * sz);   // u11
        }

        const int cn[8] = {0, 1, 2, 3, 0, 1, 2, 3};
        const int tn[8] = {1, 2, 3, 0, 2, 3, 0, 1};
        #pragma unroll
        for (int l = 0; l < 2; ++l) {
            // CNOTs: swap target amplitudes where control bit = 1 (pure register renames)
            #pragma unroll
            for (int g = 0; g < 8; ++g) {
                const int cm = 8 >> cn[g], tm = 8 >> tn[g];
                #pragma unroll
                for (int b = 0; b < 16; ++b) {
                    if (!(b & cm) || (b & tm)) continue;
                    const int bb = b | tm;
                    const float tr = sr[b], ti = si[b];
                    sr[b] = sr[bb];  si[b] = si[bb];
                    sr[bb] = tr;     si[bb] = ti;
                }
            }
            // Rot gates from precomputed LDS factors:
            // u00=(k0,-k1) u01=(-k2,-k3) u10=(k2,-k3) u11=(k0,k1)
            #pragma unroll
            for (int i = 0; i < 4; ++i) {
                const int gate = l * 4 + i;
                const float k0 = lds_rot[gate * 4 + 0];
                const float k1 = lds_rot[gate * 4 + 1];
                const float k2 = lds_rot[gate * 4 + 2];
                const float k3 = lds_rot[gate * 4 + 3];
                apply2x2(sr, si, 8 >> i,
                          k0, -k1,
                         -k2, -k3,
                          k2, -k3,
                          k0,  k1);
            }
        }

        // <Z_i> = sum_b |amp_b|^2 * (+1 if bit_i==0 else -1)
        #pragma unroll
        for (int i = 0; i < 4; ++i) {
            const int m = 8 >> i;
            float z = 0.0f;
            #pragma unroll
            for (int b = 0; b < 16; ++b) {
                const float pb = sr[b] * sr[b] + si[b] * si[b];
                z += (b & m) ? -pb : pb;
            }
            lds_q[row * 4 + i] = z;
        }
    }
    __syncthreads();

    // =============== Phase 3: GEMM2 via v_wmma_f32_16x16x4_f32 (K=4 exact) ===============
    const int k0 = half * 2;  // A f32 16x4 layout: lane half selects K pair
    v2f qa0, qa1;
    qa0[0] = lds_q[(wv * 32 + l16) * 4 + k0];
    qa0[1] = lds_q[(wv * 32 + l16) * 4 + k0 + 1];
    qa1[0] = lds_q[(wv * 32 + 16 + l16) * 4 + k0];
    qa1[1] = lds_q[(wv * 32 + 16 + l16) * 4 + k0 + 1];

    #pragma unroll
    for (int nt = 0; nt < 4; ++nt) {
        const int col = nt * 16 + l16;
        v2f bm;
        bm[0] = W2[col * 4 + k0];
        bm[1] = W2[col * 4 + k0 + 1];
        const v8f zero = {};
        v8f d0 = __builtin_amdgcn_wmma_f32_16x16x4_f32(false, qa0, false, bm, (short)0, zero, false, false);
        v8f d1 = __builtin_amdgcn_wmma_f32_16x16x4_f32(false, qa1, false, bm, (short)0, zero, false, false);
        const float bias = b2[col];
        #pragma unroll
        for (int r = 0; r < 8; ++r) {
            const long m0 = rowBlock + (long)wv * 32 + (r + 8 * half);
            out[m0 * 64 + col]        = d0[r] + bias;
            out[(m0 + 16) * 64 + col] = d1[r] + bias;
        }
    }
}

extern "C" void kernel_launch(void* const* d_in, const int* in_sizes, int n_in,
                              void* d_out, int out_size, void* d_ws, size_t ws_size,
                              hipStream_t stream) {
    const float* x  = (const float*)d_in[0];
    const float* W1 = (const float*)d_in[1];
    const float* b1 = (const float*)d_in[2];
    const float* Wq = (const float*)d_in[3];
    const float* W2 = (const float*)d_in[4];
    const float* b2 = (const float*)d_in[5];
    float* out = (float*)d_out;

    const int B = in_sizes[0] / 256;     // 131072
    dim3 grid(B / 256), block(256);
    qp_fused_kernel<<<grid, block, 0, stream>>>(x, W1, b1, Wq, W2, b2, out);
}
